// DeepLSTMDecoderLayer_73100343378178
// MI455X (gfx1250) — compile-verified, matrix operand order checked by reading, not run on hardware
//
#include <hip/hip_runtime.h>
#include <cstdint>
#include <cstddef>

// ---------------------------------------------------------------------------
// Problem constants (match reference)
// ---------------------------------------------------------------------------
constexpr int kB  = 16;
constexpr int kT  = 512;
constexpr int kS  = 512;
constexpr int kH  = 1024;
constexpr int kNH = 16;
constexpr int kHD = 64;     // head dim
constexpr int kF  = 4096;

typedef unsigned short u16;
typedef __attribute__((ext_vector_type(16))) __bf16 v16bf;
typedef __attribute__((ext_vector_type(8)))  float  v8f;

// ---------------------------------------------------------------------------
// Helpers
// ---------------------------------------------------------------------------
__device__ __forceinline__ u16 f2bf(float f) {
  union { float f; unsigned u; } u; u.f = f;
  unsigned r = u.u + 0x7FFFu + ((u.u >> 16) & 1u);   // round-to-nearest-even
  return (u16)(r >> 16);
}

__device__ __forceinline__ v16bf load_frag16(const u16* p) {
  union { v16bf v; uint4 q[2]; } u;
  const uint4* p4 = reinterpret_cast<const uint4*>(p);
  u.q[0] = p4[0];
  u.q[1] = p4[1];
  return u.v;
}

__device__ __forceinline__ v8f wmma_bf16(v16bf a, v16bf b, v8f c) {
  return __builtin_amdgcn_wmma_f32_16x16x32_bf16(
      /*neg_a=*/false, a, /*neg_b=*/false, b,
      /*c_mod=*/(short)0, c, /*reuse_a=*/false, /*reuse_b=*/false);
}

__device__ __forceinline__ float sigmoidf_(float x) {
  return 1.0f / (1.0f + __expf(-x));
}

// CDNA5 async global->LDS copy (16B per lane), tracked by ASYNCcnt.
__device__ __forceinline__ void async_copy_b128(unsigned lds_off, const u16* g) {
  asm volatile("global_load_async_to_lds_b128 %0, %1, off"
               :: "v"(lds_off), "v"((unsigned long long)(uintptr_t)g)
               : "memory");
}

// ---------------------------------------------------------------------------
// Elementwise / layout kernels
// ---------------------------------------------------------------------------
__global__ void zero_u32_kernel(unsigned* __restrict__ p, size_t n) {
  size_t i = (size_t)blockIdx.x * blockDim.x + threadIdx.x;
  if (i < n) p[i] = 0u;
}

__global__ void cvt_f32_bf16_kernel(const float* __restrict__ in,
                                    u16* __restrict__ out, size_t n) {
  size_t i = (size_t)blockIdx.x * blockDim.x + threadIdx.x;
  if (i < n) out[i] = f2bf(in[i]);
}

// out[n*K + k] = in[k*ld + n]   (weights -> bf16, B-transposed layout [N,K])
__global__ void trcvt_kernel(const float* __restrict__ in, u16* __restrict__ out,
                             int K, int N, int ld) {
  size_t i = (size_t)blockIdx.x * blockDim.x + threadIdx.x;
  size_t total = (size_t)K * N;
  if (i >= total) return;
  int k = (int)(i % K);
  int n = (int)(i / K);
  out[i] = f2bf(in[(size_t)k * ld + n]);
}

// V [B,S,H] bf16 -> VT [B,NH,HD,S] bf16  (so B-frag columns are contiguous)
__global__ void vtrans_kernel(const u16* __restrict__ V, u16* __restrict__ VT,
                              int S) {
  size_t i = (size_t)blockIdx.x * blockDim.x + threadIdx.x;
  size_t total = (size_t)kB * S * kH;
  if (i >= total) return;
  int c = (int)(i % kH);
  size_t bs = i / kH;
  int s = (int)(bs % S);
  int b = (int)(bs / S);
  int h = c >> 6;        // / kHD
  int d = c & 63;        // % kHD
  VT[(((size_t)b * kNH + h) * kHD + d) * S + s] = V[i];
}

// inp[B*T, 2H] bf16 = concat(xb, bf16(c))
__global__ void concat_kernel(const u16* __restrict__ xb,
                              const float* __restrict__ c,
                              u16* __restrict__ inp) {
  size_t i = (size_t)blockIdx.x * blockDim.x + threadIdx.x;
  size_t total = (size_t)kB * kT * 2 * kH;
  if (i >= total) return;
  int col = (int)(i % (2 * kH));
  size_t row = i / (2 * kH);
  inp[i] = (col < kH) ? xb[row * kH + col] : f2bf(c[row * kH + (col - kH)]);
}

// ---------------------------------------------------------------------------
// Block-tiled bf16 WMMA GEMM with async-to-LDS B staging.
//   C[M,N] = epilogue( A[M,K] @ BT[N,K]^T + bias )
// Block = 256 threads (8 waves) covering a 128x64 tile. Per 32-wide k-step the
// shared 64x32 B tile is staged once via global_load_async_to_lds_b128
// (double-buffered, ASYNCcnt-pipelined); each wave reads its B fragments from
// LDS (8x reuse) and streams its private A fragments from global.
// ---------------------------------------------------------------------------
enum { OUT_F32 = 0, OUT_BF16 = 1, OUT_F32_ADD = 2, OUT_BF16_RELU = 3 };

__global__ __launch_bounds__(256) void gemm_lds_kernel(
    const u16* __restrict__ A, const u16* __restrict__ BT,
    const float* __restrict__ bias, void* __restrict__ Cout,
    int M, int N, int K, float scale, int mode) {
  constexpr int RS = 40;                       // LDS row stride in u16 (80B)
  __shared__ __align__(16) u16 sB[2][64 * RS];

  const int tid  = threadIdx.x;
  const int lane = tid & 31;
  const int wid  = tid >> 5;                   // 0..7
  const int tilesN = N >> 6;
  const int bm = blockIdx.x / tilesN;
  const int bn = blockIdx.x % tilesN;
  const int n0 = bn * 64;
  const int m0 = bm * 128 + wid * 16;
  const int hi = lane >> 4, lo = lane & 15;

  // staging assignment: thread t copies 16B: B row = t>>2, 16B segment = t&3
  const int srow = tid >> 2, sseg = tid & 3;
  const u16* gB = BT + (size_t)(n0 + srow) * K + sseg * 8;
  const unsigned ldsB0 =
      (unsigned)(uintptr_t)(&sB[0][srow * RS + sseg * 8]);
  const unsigned ldsB1 =
      (unsigned)(uintptr_t)(&sB[1][srow * RS + sseg * 8]);

  const u16* ap = A + (size_t)(m0 + lo) * K + hi * 16;

  v8f acc[4] = {{}, {}, {}, {}};

  async_copy_b128(ldsB0, gB);                  // prologue: stage k-step 0
  const int ksteps = K >> 5;
  for (int ks = 0; ks < ksteps; ++ks) {
    const int k0 = ks << 5;
    if (ks + 1 < ksteps) {                     // issue next stage, keep 1 in flight
      async_copy_b128((ks & 1) ? ldsB0 : ldsB1, gB + k0 + 32);
      asm volatile("s_wait_asynccnt 0x1" ::: "memory");
    } else {
      asm volatile("s_wait_asynccnt 0x0" ::: "memory");
    }
    __syncthreads();

    const u16* bbuf = (ks & 1) ? &sB[1][0] : &sB[0][0];
    v16bf a = load_frag16(ap + k0);
    acc[0] = wmma_bf16(a, load_frag16(bbuf + ( 0 + lo) * RS + hi * 16), acc[0]);
    acc[1] = wmma_bf16(a, load_frag16(bbuf + (16 + lo) * RS + hi * 16), acc[1]);
    acc[2] = wmma_bf16(a, load_frag16(bbuf + (32 + lo) * RS + hi * 16), acc[2]);
    acc[3] = wmma_bf16(a, load_frag16(bbuf + (48 + lo) * RS + hi * 16), acc[3]);
    __syncthreads();                           // protect buffer for next overwrite
  }

  float* cf = (float*)Cout;
  u16*   cb = (u16*)Cout;
  for (int nt = 0; nt < 4; ++nt) {
    const int n = n0 + nt * 16 + lo;
    const float bv = bias ? bias[n] : 0.0f;
    for (int r = 0; r < 8; ++r) {
      const int m = m0 + r + hi * 8;
      const float v = (acc[nt][r] + bv) * scale;
      const size_t idx = (size_t)m * N + n;
      if (mode == OUT_F32)            cf[idx]  = v;
      else if (mode == OUT_F32_ADD)   cf[idx] += v;
      else if (mode == OUT_BF16)      cb[idx]  = f2bf(v);
      else                            cb[idx]  = f2bf(fmaxf(v, 0.0f));
    }
  }
}

// ---------------------------------------------------------------------------
// Flash attention: one wave per (b, head, 16-query tile).
// Q pre-scaled by HD^-0.5. K as bf16 [B,Skv,H]; V as VT bf16 [B,NH,HD,Skv].
// Online softmax with shfl_xor row reductions; P re-fragmented through LDS.
// ---------------------------------------------------------------------------
__global__ __launch_bounds__(32) void flash_attn_kernel(
    const u16* __restrict__ Q, const u16* __restrict__ Kb,
    const u16* __restrict__ VT, const float* __restrict__ bias,
    int causal, int Tq, int Skv, u16* __restrict__ O) {
  __shared__ float plds[16][32];
  const int lane = threadIdx.x & 31;
  const int hi = lane >> 4, lo = lane & 15;

  int wg = blockIdx.x;
  const int qtiles = Tq >> 4;
  const int qt = wg % qtiles; wg /= qtiles;
  const int h  = wg % kNH;
  const int b  = wg / kNH;
  const int t0 = qt * 16;

  const u16* qbase = Q + ((size_t)b * Tq + t0 + lo) * kH + h * kHD + hi * 16;
  const v16bf aq0 = load_frag16(qbase);
  const v16bf aq1 = load_frag16(qbase + 32);

  float rm[8], rl[8];
  for (int r = 0; r < 8; ++r) { rm[r] = -3.0e38f; rl[r] = 0.0f; }
  v8f oacc[4] = {{}, {}, {}, {}};

  for (int s0 = 0; s0 < Skv; s0 += 32) {
    if (causal && s0 > t0 + 15) break;

    // ---- logits: 16x32 tile as two 16x16 WMMA accumulations over d=64 ----
    v8f lg0 = {}, lg1 = {};
    {
      const u16* kb0 = Kb + ((size_t)b * Skv + s0 + lo) * kH + h * kHD + hi * 16;
      const u16* kb1 = kb0 + (size_t)16 * kH;
      lg0 = wmma_bf16(aq0, load_frag16(kb0),      lg0);
      lg0 = wmma_bf16(aq1, load_frag16(kb0 + 32), lg0);
      lg1 = wmma_bf16(aq0, load_frag16(kb1),      lg1);
      lg1 = wmma_bf16(aq1, load_frag16(kb1 + 32), lg1);
    }

    const float bv0 = bias ? bias[(size_t)b * Skv + s0 + lo]      : 0.0f;
    const float bv1 = bias ? bias[(size_t)b * Skv + s0 + 16 + lo] : 0.0f;
    const int sn0 = s0 + lo, sn1 = s0 + 16 + lo;

    // ---- online softmax update for this 16x32 block ----
    for (int r = 0; r < 8; ++r) {
      const int m = t0 + r + hi * 8;
      float v0 = lg0[r] + bv0;
      float v1 = lg1[r] + bv1;
      if (causal) {
        if (sn0 > m) v0 = -1.0e30f;
        if (sn1 > m) v1 = -1.0e30f;
      }
      float x = fmaxf(v0, v1);
      for (int msk = 8; msk; msk >>= 1) x = fmaxf(x, __shfl_xor(x, msk, 32));
      const float nm = fmaxf(rm[r], x);
      const float cr = __expf(rm[r] - nm);
      rm[r] = nm;
      rl[r] *= cr;
      const float p0 = __expf(v0 - nm);
      const float p1 = __expf(v1 - nm);
      float sr = p0 + p1;
      for (int msk = 8; msk; msk >>= 1) sr += __shfl_xor(sr, msk, 32);
      rl[r] += sr;
      plds[r + hi * 8][lo]      = p0;
      plds[r + hi * 8][16 + lo] = p1;
      oacc[0][r] *= cr; oacc[1][r] *= cr; oacc[2][r] *= cr; oacc[3][r] *= cr;
    }
    __syncthreads();

    // ---- re-fragment P (C-layout -> A-layout) through LDS ----
    union { v16bf v; u16 s[16]; } pu;
    {
      const float* prow = &plds[lo][hi * 16];
      for (int i = 0; i < 16; ++i) pu.s[i] = f2bf(prow[i]);
    }

    // ---- O += P @ V_tile (four 16-wide d-tiles) ----
    const u16* vbase = VT + (((size_t)b * kNH + h) * kHD + lo) * Skv + s0 + hi * 16;
    oacc[0] = wmma_bf16(pu.v, load_frag16(vbase),                    oacc[0]);
    oacc[1] = wmma_bf16(pu.v, load_frag16(vbase + (size_t)16 * Skv), oacc[1]);
    oacc[2] = wmma_bf16(pu.v, load_frag16(vbase + (size_t)32 * Skv), oacc[2]);
    oacc[3] = wmma_bf16(pu.v, load_frag16(vbase + (size_t)48 * Skv), oacc[3]);
    __syncthreads();
  }

  // ---- normalize and write bf16 output tile ----
  for (int nt = 0; nt < 4; ++nt) {
    const int d = h * kHD + nt * 16 + lo;
    for (int r = 0; r < 8; ++r) {
      const int m = t0 + r + hi * 8;
      O[((size_t)b * Tq + m) * kH + d] = f2bf(oacc[nt][r] / rl[r]);
    }
  }
}

// ---------------------------------------------------------------------------
// Fused LSTM step: g = Gx_t + h_{t-1} @ Wh; gate math; writes h_t (bf16), c (f32).
// Grid = 64 waves; each wave owns hidden columns [n0, n0+16) across all 4 gates.
// ---------------------------------------------------------------------------
__global__ __launch_bounds__(32) void lstm_step_kernel(
    const u16* __restrict__ hprev, long hstride,
    const u16* __restrict__ WhT,              // [4H, H] bf16 (B-transposed)
    const float* __restrict__ Gx, long gstride,
    float* __restrict__ cstate,               // [B, H] f32
    u16* __restrict__ hout, long hostride) {
  const int lane = threadIdx.x & 31;
  const int hi = lane >> 4, lo = lane & 15;
  const int n0 = blockIdx.x * 16;

  v8f acc[4] = {{}, {}, {}, {}};
  const u16* ap = hprev + (size_t)lo * hstride + hi * 16;
  const u16* bp[4];
  for (int g = 0; g < 4; ++g)
    bp[g] = WhT + (size_t)(g * kH + n0 + lo) * kH + hi * 16;

  for (int k0 = 0; k0 < kH; k0 += 32) {
    v16bf a = load_frag16(ap + k0);
    acc[0] = wmma_bf16(a, load_frag16(bp[0] + k0), acc[0]);
    acc[1] = wmma_bf16(a, load_frag16(bp[1] + k0), acc[1]);
    acc[2] = wmma_bf16(a, load_frag16(bp[2] + k0), acc[2]);
    acc[3] = wmma_bf16(a, load_frag16(bp[3] + k0), acc[3]);
  }

  const int n = n0 + lo;
  for (int r = 0; r < 8; ++r) {
    const int m = r + hi * 8;                  // batch index
    const float* grow = Gx + (size_t)m * gstride;
    const float gi = acc[0][r] + grow[n];
    const float gj = acc[1][r] + grow[kH + n];
    const float gf = acc[2][r] + grow[2 * kH + n];
    const float go = acc[3][r] + grow[3 * kH + n];
    const float cn = sigmoidf_(gf) * cstate[(size_t)m * kH + n] +
                     sigmoidf_(gi) * tanhf(gj);
    const float hn = sigmoidf_(go) * tanhf(cn);
    cstate[(size_t)m * kH + n] = cn;
    hout[(size_t)m * hostride + n] = f2bf(hn);
  }
}

// ---------------------------------------------------------------------------
// Host-side helpers
// ---------------------------------------------------------------------------
static inline size_t align256(size_t x) { return (x + 255) & ~(size_t)255; }

static void launch_gemm(const u16* A, const u16* BT, const float* bias, void* C,
                        int M, int N, int K, float scale, int mode,
                        hipStream_t s) {
  const int blocks = (M / 128) * (N / 64);
  gemm_lds_kernel<<<blocks, 256, 0, s>>>(A, BT, bias, C, M, N, K, scale, mode);
}

static void launch_trcvt(const float* in, u16* out, int K, int N, int ld,
                         hipStream_t s) {
  const size_t cnt = (size_t)K * N;
  trcvt_kernel<<<(unsigned)((cnt + 255) / 256), 256, 0, s>>>(in, out, K, N, ld);
}

static void launch_cvt(const float* in, u16* out, size_t cnt, hipStream_t s) {
  cvt_f32_bf16_kernel<<<(unsigned)((cnt + 255) / 256), 256, 0, s>>>(in, out, cnt);
}

extern "C" void kernel_launch(void* const* d_in, const int* in_sizes, int n_in,
                              void* d_out, int out_size, void* d_ws,
                              size_t ws_size, hipStream_t stream) {
  (void)in_sizes; (void)n_in; (void)out_size; (void)ws_size;

  // ---- inputs (setup_inputs dict order) ----
  const float* x        = (const float*)d_in[0];
  const float* memory   = (const float*)d_in[1];
  const float* src_bias = (const float*)d_in[2];
  // d_in[3] = tgt_bias (causal mask) -- applied analytically in flash kernel
  const float* w[8]; const float* bw[8];
  for (int i = 0; i < 8; ++i) { w[i] = (const float*)d_in[4 + 2 * i];
                                bw[i] = (const float*)d_in[5 + 2 * i]; }
  // w/bw order: src_wq, src_wk, src_wv, src_wo, tgt_wq, tgt_wk, tgt_wv, tgt_wo
  const float* lstm_w = (const float*)d_in[20];
  const float* lstm_b = (const float*)d_in[21];
  const float* ffn_w1 = (const float*)d_in[22];
  const float* ffn_b1 = (const float*)d_in[23];
  const float* ffn_w2 = (const float*)d_in[24];
  const float* ffn_b2 = (const float*)d_in[25];
  float* out = (float*)d_out;

  // ---- workspace layout ----
  char* wp = (char*)d_ws;
  size_t off = 0;
  auto alloc = [&](size_t bytes) -> void* {
    void* p = wp + off; off = align256(off + bytes); return p;
  };
  const size_t szBTH  = (size_t)kB * kT * kH;   // == B*S*H
  u16* wT[8];
  for (int i = 0; i < 8; ++i) wT[i] = (u16*)alloc((size_t)kH * kH * 2);
  u16* WxT = (u16*)alloc((size_t)4 * kH * 2 * kH * 2);     // [4H, 2H]
  u16* WhT = (u16*)alloc((size_t)4 * kH * kH * 2);         // [4H, H]
  u16* W1T = (u16*)alloc((size_t)kF * kH * 2);             // [F, H]
  u16* W2T = (u16*)alloc((size_t)kH * kF * 2);             // [H, F]
  u16* xb      = (u16*)alloc(szBTH * 2);
  u16* memb    = (u16*)alloc(szBTH * 2);
  u16* Qb      = (u16*)alloc(szBTH * 2);
  u16* Kbuf    = (u16*)alloc(szBTH * 2);
  u16* Vbuf    = (u16*)alloc(szBTH * 2);
  u16* VTb     = (u16*)alloc(szBTH * 2);
  u16* attnO   = (u16*)alloc(szBTH * 2);
  float* c_f32 = (float*)alloc(szBTH * 4);
  u16* inp     = (u16*)alloc(szBTH * 2 * 2);               // [B,T,2H]
  float* Gx    = (float*)alloc((size_t)kB * kT * 4 * kH * 4);
  u16* h_all   = (u16*)alloc(szBTH * 2);
  u16* hzero   = (u16*)alloc((size_t)kB * kH * 2);
  float* cst   = (float*)alloc((size_t)kB * kH * 4);
  u16* mid     = (u16*)alloc((size_t)kB * kT * kF * 2);

  const int M = kB * kT;            // 8192
  const float qscale = 0.125f;      // HD^-0.5

  // ---- stage 0: weight/activation conversion to bf16 ----
  for (int i = 0; i < 8; ++i) launch_trcvt(w[i], wT[i], kH, kH, kH, stream);
  launch_trcvt(lstm_w,              WxT, 2 * kH, 4 * kH, 4 * kH, stream);
  launch_trcvt(lstm_w + (size_t)2 * kH * 4 * kH, WhT, kH, 4 * kH, 4 * kH, stream);
  launch_trcvt(ffn_w1, W1T, kH, kF, kF, stream);
  launch_trcvt(ffn_w2, W2T, kF, kH, kH, stream);
  launch_cvt(x,      xb,   szBTH, stream);
  launch_cvt(memory, memb, szBTH, stream);

  // ---- stage 1: source (cross) attention ----
  launch_gemm(xb,   wT[0], bw[0], Qb,   M, kH, kH, qscale, OUT_BF16, stream);
  launch_gemm(memb, wT[1], bw[1], Kbuf, M, kH, kH, 1.0f,   OUT_BF16, stream);
  launch_gemm(memb, wT[2], bw[2], Vbuf, M, kH, kH, 1.0f,   OUT_BF16, stream);
  vtrans_kernel<<<(unsigned)((szBTH + 255) / 256), 256, 0, stream>>>(Vbuf, VTb, kS);
  flash_attn_kernel<<<kB * kNH * (kT / 16), 32, 0, stream>>>(
      Qb, Kbuf, VTb, src_bias, /*causal=*/0, kT, kS, attnO);
  launch_gemm(attnO, wT[3], bw[3], c_f32, M, kH, kH, 1.0f, OUT_F32, stream);

  // ---- stage 2: target (causal self) attention, accumulated into c ----
  launch_gemm(xb, wT[4], bw[4], Qb,   M, kH, kH, qscale, OUT_BF16, stream);
  launch_gemm(xb, wT[5], bw[5], Kbuf, M, kH, kH, 1.0f,   OUT_BF16, stream);
  launch_gemm(xb, wT[6], bw[6], Vbuf, M, kH, kH, 1.0f,   OUT_BF16, stream);
  vtrans_kernel<<<(unsigned)((szBTH + 255) / 256), 256, 0, stream>>>(Vbuf, VTb, kT);
  flash_attn_kernel<<<kB * kNH * (kT / 16), 32, 0, stream>>>(
      Qb, Kbuf, VTb, nullptr, /*causal=*/1, kT, kT, attnO);
  launch_gemm(attnO, wT[7], bw[7], c_f32, M, kH, kH, 1.0f, OUT_F32_ADD, stream);

  // ---- stage 3: LSTM input + hoisted Gx GEMM ----
  {
    const size_t cnt = szBTH * 2;
    concat_kernel<<<(unsigned)((cnt + 255) / 256), 256, 0, stream>>>(xb, c_f32, inp);
  }
  launch_gemm(inp, WxT, lstm_b, Gx, M, 4 * kH, 2 * kH, 1.0f, OUT_F32, stream);

  // ---- stage 4: recurrent LSTM (512 fused step kernels) ----
  {
    const size_t nz1 = (size_t)kB * kH / 2;   // hzero bytes/4
    zero_u32_kernel<<<(unsigned)((nz1 + 255) / 256), 256, 0, stream>>>(
        (unsigned*)hzero, nz1);
    const size_t nz2 = (size_t)kB * kH;       // cst f32 count
    zero_u32_kernel<<<(unsigned)((nz2 + 255) / 256), 256, 0, stream>>>(
        (unsigned*)cst, nz2);
  }
  for (int t = 0; t < kT; ++t) {
    const u16* hprev = (t == 0) ? hzero : (h_all + (size_t)(t - 1) * kH);
    const long hstride = (t == 0) ? (long)kH : (long)kT * kH;
    lstm_step_kernel<<<kH / 16, 32, 0, stream>>>(
        hprev, hstride, WhT,
        Gx + (size_t)t * 4 * kH, (long)kT * 4 * kH,
        cst, h_all + (size_t)t * kH, (long)kT * kH);
  }

  // ---- stage 5: FFN ----
  launch_gemm(h_all, W1T, ffn_b1, mid, M, kF, kH, 1.0f, OUT_BF16_RELU, stream);
  launch_gemm(mid,   W2T, ffn_b2, out, M, kH, kF, 1.0f, OUT_F32, stream);
}
